// CAGenerator_11845519802531
// MI455X (gfx1250) — matrix operand order repaired and checked by reading, not compiled, not run on hardware
//
#include <hip/hip_runtime.h>

typedef __attribute__((ext_vector_type(16))) __bf16          v16bf;
typedef __attribute__((ext_vector_type(8)))  float           v8f;
typedef __attribute__((ext_vector_type(4)))  float           v4f;
typedef __attribute__((ext_vector_type(4)))  unsigned        v4u;
typedef __attribute__((ext_vector_type(2)))  unsigned        v2u;

#define NSTATE 20
#define HID    128
#define NPIX   1024
#define STEPS  64
#define THRESH 0.1f
#define KPERC  180            // 20 ch * 9 taps
#define SPITCH 34             // shadow: 32 + halo
#define SCH    32             // shadow channels padded 20 -> 32
#define NWAVES 8
#define NTILES 64             // 1024 px / 16

// ---------------- LDS layout (bytes), all 32B aligned ----------------
#define OFF_STATE   0
#define SZ_STATE    (NPIX*NSTATE*4)                   // 81920  f32 master [p][ch]
#define OFF_SH      (OFF_STATE + SZ_STATE)
#define SZ_SH       (SPITCH*SPITCH*SCH*2)             // 73984  bf16 shadow [y][x][ch]
#define OFF_PERCFR  (OFF_SH + SZ_SH)
#define SZ_PERCFR   (9*8*512*2)                       // 73728  A-frags per tap x mt
#define OFF_UP1FR   (OFF_PERCFR + SZ_PERCFR)
#define SZ_UP1FR    (32*512*2)                        // 32768
#define OFF_UP2FR   (OFF_UP1FR + SZ_UP1FR)
#define SZ_UP2FR    (8*512*2)                         // 8192
#define OFF_SCR     (OFF_UP2FR + SZ_UP2FR)
#define SZ_SCR      (NWAVES*2048*2)                   // 32768  per-wave 16n x 128k bf16
#define OFF_PRE     (OFF_SCR + SZ_SCR)
#define SZ_PRE      1024
#define OFF_BP      (OFF_PRE + SZ_PRE)
#define SZ_BP       (128*4)
#define OFF_BU1     (OFF_BP + SZ_BP)
#define SZ_BU1      (128*4)
#define OFF_BU2     (OFF_BU1 + SZ_BU1)
#define SZ_BU2      (32*4)
#define OFF_H50     (OFF_BU2 + SZ_BU2)
#define SZ_H50      (64*4)
#define SMEM_BYTES  (OFF_H50 + SZ_H50)                // 305792 B <= 320 KB/WGP

// --- bf16 helpers (RNE) ---
__device__ __forceinline__ unsigned short f2bf(float f) {
  union { float f; unsigned u; } c; c.f = f;
  unsigned u = c.u;
  u += 0x7FFFu + ((u >> 16) & 1u);
  return (unsigned short)(u >> 16);
}
__device__ __forceinline__ unsigned pack2bf(float a, float b) {
  return (unsigned)f2bf(a) | ((unsigned)f2bf(b) << 16);
}

// A-fragment k order used here (contiguous, matches B): lanes0-15 K{0-15}, lanes16-31 K{16-31}
__device__ __forceinline__ int a_k(int lane, int slot) {
  return ((lane >> 4) << 4) + slot;
}
// B k order used here, same as A: lanes0-15 hold K=0..15, lanes16-31 K=16..31 (slot s -> K = s + 16h)
// => a 16-half CONTIGUOUS K-run per lane; LDS is laid out so these are b128 loads.

__device__ __forceinline__ v16bf ldv(const unsigned short* p) {
  return *(const v16bf*)p;
}
__device__ __forceinline__ v8f wmma_bf16(v16bf a, v16bf b, v8f c) {
  return __builtin_amdgcn_wmma_f32_16x16x32_bf16(false, a, false, b, (short)0, c,
                                                 false, false);
}

// up1 -> up2 tail for one pixel-tile: perc acc -> bounce -> relu(W1..) -> W2 -> state+=
__device__ __forceinline__ void tile_tail(v8f (&acc)[8], int p, int n, int lane,
                                          int khalf, int mhalf,
                                          unsigned short* ws,
                                          const unsigned short* up1Fr,
                                          const unsigned short* up2Fr,
                                          float* stateP, const float* bpl,
                                          const float* bu1l, const float* bu2l)
{
  // perceive bias + pack -> transposed bounce buffer (one b128 store per mt)
  #pragma unroll
  for (int mt = 0; mt < 8; ++mt) {
    int m0 = mt * 16 + mhalf;
    v4u pk;
    #pragma unroll
    for (int d = 0; d < 4; ++d)
      pk[d] = pack2bf(acc[mt][2 * d] + bpl[m0 + 2 * d],
                      acc[mt][2 * d + 1] + bpl[m0 + 2 * d + 1]);
    *(v4u*)(ws + (n << 7) + m0) = pk;
  }

  // up1: relu(W1 @ perc + b1)
  v16bf Bh[4];
  #pragma unroll
  for (int kt = 0; kt < 4; ++kt)
    Bh[kt] = ldv(ws + (n << 7) + kt * 32 + khalf);
  #pragma unroll
  for (int mt = 0; mt < 8; ++mt) {
    v8f a = (v8f){};
    #pragma unroll
    for (int kt = 0; kt < 4; ++kt)
      a = wmma_bf16(ldv(up1Fr + ((mt * 4 + kt) << 9) + lane * 16), Bh[kt], a);
    acc[mt] = a;
  }
  #pragma unroll
  for (int mt = 0; mt < 8; ++mt) {
    int m0 = mt * 16 + mhalf;
    v4u pk;
    #pragma unroll
    for (int d = 0; d < 4; ++d)
      pk[d] = pack2bf(fmaxf(acc[mt][2 * d] + bu1l[m0 + 2 * d], 0.f),
                      fmaxf(acc[mt][2 * d + 1] + bu1l[m0 + 2 * d + 1], 0.f));
    *(v4u*)(ws + (n << 7) + m0) = pk;
  }

  // up2: W2 @ hidden + b2, vectorized accumulate into f32 master [p][ch]
  v16bf Bu[4];
  #pragma unroll
  for (int kt = 0; kt < 4; ++kt)
    Bu[kt] = ldv(ws + (n << 7) + kt * 32 + khalf);

  {  // mt = 0: channels mhalf..mhalf+7 (all valid)
    v8f a = (v8f){};
    #pragma unroll
    for (int kt = 0; kt < 4; ++kt)
      a = wmma_bf16(ldv(up2Fr + (kt << 9) + lane * 16), Bu[kt], a);
    float* sp = stateP + p * NSTATE + mhalf;
    v4f s0 = *(v4f*)sp, s1 = *(v4f*)(sp + 4);
    #pragma unroll
    for (int d = 0; d < 4; ++d) {
      s0[d] += a[d] + bu2l[mhalf + d];
      s1[d] += a[4 + d] + bu2l[mhalf + 4 + d];
    }
    *(v4f*)sp = s0; *(v4f*)(sp + 4) = s1;
  }
  {  // mt = 1: only lanes with mhalf==0 carry valid channels 16..19
    v8f a = (v8f){};
    #pragma unroll
    for (int kt = 0; kt < 4; ++kt)
      a = wmma_bf16(ldv(up2Fr + ((4 + kt) << 9) + lane * 16), Bu[kt], a);
    if (mhalf == 0) {
      float* sp = stateP + p * NSTATE + 16;
      v4f s = *(v4f*)sp;
      #pragma unroll
      for (int d = 0; d < 4; ++d) s[d] += a[d] + bu2l[16 + d];
      *(v4f*)sp = s;
    }
  }
}

extern __shared__ char smem[];

__global__ __launch_bounds__(256, 1)
void nca_persist_kernel(const float* __restrict__ z,
                        const float* __restrict__ w1,  const float* __restrict__ b1,
                        const float* __restrict__ w2,  const float* __restrict__ b2,
                        const float* __restrict__ wp,  const float* __restrict__ bp,
                        const float* __restrict__ wu1, const float* __restrict__ bu1,
                        const float* __restrict__ wu2, const float* __restrict__ bu2,
                        float* __restrict__ out)
{
  const int tid  = threadIdx.x;
  const int lane = tid & 31;
  const int wave = tid >> 5;
  const int img  = blockIdx.x;

  float*          stateP = (float*)(smem + OFF_STATE);           // [p][ch]
  unsigned short* sh     = (unsigned short*)(smem + OFF_SH);     // [(y+1)*34+(x+1)][ch]
  unsigned short* percFr = (unsigned short*)(smem + OFF_PERCFR);
  unsigned short* up1Fr  = (unsigned short*)(smem + OFF_UP1FR);
  unsigned short* up2Fr  = (unsigned short*)(smem + OFF_UP2FR);
  unsigned short* scr    = (unsigned short*)(smem + OFF_SCR);
  unsigned char*  pre    = (unsigned char*)(smem + OFF_PRE);
  float*          bpl    = (float*)(smem + OFF_BP);
  float*          bu1l   = (float*)(smem + OFF_BU1);
  float*          bu2l   = (float*)(smem + OFF_BU2);
  float*          h50    = (float*)(smem + OFF_H50);

  // warm L2 with the weights (global_prefetch_b8 path)
  for (int i = tid * 64; i < KPERC * HID * 4; i += 256 * 64)
    __builtin_prefetch((const char*)wp + i, 0, 3);
  for (int i = tid * 64; i < HID * HID * 4; i += 256 * 64)
    __builtin_prefetch((const char*)wu1 + i, 0, 3);

  // ---- one-time: pack weights into WMMA A-fragments (bf16, exact lane layout) ----
  for (int i = tid; i < 9 * 8 * 512; i += 256) {         // perceive: 9 taps x 8 mt
    int frag = i >> 9, e = i & 511, l = e >> 4, s = e & 15;
    int tap = frag >> 3, mt = frag & 7;
    int ch = a_k(l, s);
    int m  = mt * 16 + (l & 15);
    percFr[i] = f2bf(ch < NSTATE ? wp[m * KPERC + ch * 9 + tap] : 0.f);
  }
  for (int i = tid; i < 32 * 512; i += 256) {            // up1 128x128
    int frag = i >> 9, e = i & 511, l = e >> 4, s = e & 15;
    int mt = frag >> 2, kt = frag & 3;
    int k = kt * 32 + a_k(l, s);
    int m = mt * 16 + (l & 15);
    up1Fr[i] = f2bf(wu1[m * HID + k]);
  }
  for (int i = tid; i < 8 * 512; i += 256) {             // up2 20x128 (M pad 32)
    int frag = i >> 9, e = i & 511, l = e >> 4, s = e & 15;
    int mt = frag >> 2, kt = frag & 3;
    int k = kt * 32 + a_k(l, s);
    int m = mt * 16 + (l & 15);
    up2Fr[i] = f2bf(m < NSTATE ? wu2[m * HID + k] : 0.f);
  }
  for (int i = tid; i < NPIX * NSTATE; i += 256) stateP[i] = 0.f;
  for (int i = tid; i < SPITCH * SPITCH * SCH / 2; i += 256) ((unsigned*)sh)[i] = 0u;
  if (tid < 128) { bpl[tid] = bp[tid]; bu1l[tid] = bu1[tid]; }
  if (tid < 32)  bu2l[tid] = (tid < NSTATE) ? bu2[tid] : 0.f;
  __syncthreads();

  // ---- init_state: 1x1 conv -> relu -> 1x1 conv at center pixel (16,16) ----
  if (tid < 50) {
    float a = b1[tid];
    for (int k = 0; k < 100; ++k) a += w1[tid * 100 + k] * z[img * 100 + k];
    h50[tid] = fmaxf(a, 0.f);
  }
  __syncthreads();
  const int cPix = 16 * 32 + 16;
  const int cSh  = (17 * SPITCH + 17) * SCH;
  if (tid < 19) {
    float a = b2[tid];
    for (int h = 0; h < 50; ++h) a += w2[tid * 50 + h] * h50[h];
    stateP[cPix * NSTATE + tid + 1] = a;
    sh[cSh + tid + 1] = f2bf(a);
  }
  if (tid == 255) { stateP[cPix * NSTATE] = 0.5f; sh[cSh] = f2bf(0.5f); }
  __syncthreads();

  unsigned short* ws = scr + wave * 2048;    // per-wave bounce: idx = n*128 + k (bf16)
  const int n     = lane & 15;
  const int khalf = (lane >> 4) << 4;        // B-fragment K offset for this lane half
  const int mhalf = (lane >> 4) << 3;        // D-fragment M offset for this lane half

  for (int step = 0; step < STEPS; ++step) {
    // ---- phase 0: pre alive mask = maxpool3(ch0) > T (clamped window) ----
    for (int p = tid; p < NPIX; p += 256) {
      int y = p >> 5, x = p & 31;
      float mx = -1e30f;
      for (int dy = -1; dy <= 1; ++dy) {
        int yy = y + dy; if ((unsigned)yy >= 32u) continue;
        for (int dx = -1; dx <= 1; ++dx) {
          int xx = x + dx; if ((unsigned)xx >= 32u) continue;
          mx = fmaxf(mx, stateP[((yy << 5) + xx) * NSTATE]);
        }
      }
      pre[p] = (mx > THRESH) ? 1 : 0;
    }
    __syncthreads();

    // ---- phase A: two tiles per wave; each perceive A-frag load feeds 2 WMMAs ----
    #pragma unroll 1
    for (int i = 0; i < 4; ++i) {
      const int tA = wave + i * 16, tB = tA + 8;
      const int pA = tA * 16 + n, pB = tB * 16 + n;
      const int yA = pA >> 5, xA = pA & 31;
      const int yB = pB >> 5, xB = pB & 31;

      v8f accA[8], accB[8];
      #pragma unroll
      for (int mt = 0; mt < 8; ++mt) { accA[mt] = (v8f){}; accB[mt] = (v8f){}; }

      #pragma unroll
      for (int tap = 0; tap < 9; ++tap) {
        int dy = tap / 3, dx = tap - dy * 3;
        v16bf BA = ldv(sh + ((yA + dy) * SPITCH + (xA + dx)) * SCH + khalf);
        v16bf BB = ldv(sh + ((yB + dy) * SPITCH + (xB + dx)) * SCH + khalf);
        #pragma unroll
        for (int mt = 0; mt < 8; ++mt) {
          v16bf A = ldv(percFr + ((tap * 8 + mt) << 9) + lane * 16);
          accA[mt] = wmma_bf16(A, BA, accA[mt]);
          accB[mt] = wmma_bf16(A, BB, accB[mt]);
        }
      }
      tile_tail(accA, pA, n, lane, khalf, mhalf, ws, up1Fr, up2Fr,
                stateP, bpl, bu1l, bu2l);
      tile_tail(accB, pB, n, lane, khalf, mhalf, ws, up1Fr, up2Fr,
                stateP, bpl, bu1l, bu2l);
    }
    __syncthreads();

    // ---- phase B1: post alive mask on updated ch0; alive = pre & post ----
    for (int p = tid; p < NPIX; p += 256) {
      int y = p >> 5, x = p & 31;
      float mx = -1e30f;
      for (int dy = -1; dy <= 1; ++dy) {
        int yy = y + dy; if ((unsigned)yy >= 32u) continue;
        for (int dx = -1; dx <= 1; ++dx) {
          int xx = x + dx; if ((unsigned)xx >= 32u) continue;
          mx = fmaxf(mx, stateP[((yy << 5) + xx) * NSTATE]);
        }
      }
      pre[p] = pre[p] && (mx > THRESH);
    }
    __syncthreads();

    // ---- phase B2: apply mask + rebuild bf16 shadow (vectorized) ----
    for (int p = tid; p < NPIX; p += 256) {
      int y = p >> 5, x = p & 31;
      float alive = pre[p] ? 1.f : 0.f;
      float* sp = stateP + p * NSTATE;
      int shb = ((y + 1) * SPITCH + (x + 1)) * SCH;
      v4f a0 = *(v4f*)(sp)      * alive;
      v4f a1 = *(v4f*)(sp + 4)  * alive;
      v4f a2 = *(v4f*)(sp + 8)  * alive;
      v4f a3 = *(v4f*)(sp + 12) * alive;
      v4f a4 = *(v4f*)(sp + 16) * alive;
      *(v4f*)(sp)      = a0;
      *(v4f*)(sp + 4)  = a1;
      *(v4f*)(sp + 8)  = a2;
      *(v4f*)(sp + 12) = a3;
      *(v4f*)(sp + 16) = a4;
      v4u p0, p1; v2u p2;
      p0[0] = pack2bf(a0[0], a0[1]); p0[1] = pack2bf(a0[2], a0[3]);
      p0[2] = pack2bf(a1[0], a1[1]); p0[3] = pack2bf(a1[2], a1[3]);
      p1[0] = pack2bf(a2[0], a2[1]); p1[1] = pack2bf(a2[2], a2[3]);
      p1[2] = pack2bf(a3[0], a3[1]); p1[3] = pack2bf(a3[2], a3[3]);
      p2[0] = pack2bf(a4[0], a4[1]); p2[1] = pack2bf(a4[2], a4[3]);
      *(v4u*)(sh + shb)      = p0;
      *(v4u*)(sh + shb + 8)  = p1;
      *(v2u*)(sh + shb + 16) = p2;
    }
    __syncthreads();
  }

  // ---- output: channel 0 only (NC=1) ----
  for (int p = tid; p < NPIX; p += 256)
    out[img * NPIX + p] = stateP[p * NSTATE];
}

extern "C" void kernel_launch(void* const* d_in, const int* in_sizes, int n_in,
                              void* d_out, int out_size, void* d_ws, size_t ws_size,
                              hipStream_t stream) {
  (void)d_ws; (void)ws_size; (void)n_in; (void)out_size;
  const float* z   = (const float*)d_in[0];
  const float* w1  = (const float*)d_in[1];
  const float* b1  = (const float*)d_in[2];
  const float* w2  = (const float*)d_in[3];
  const float* b2  = (const float*)d_in[4];
  const float* wp  = (const float*)d_in[5];
  const float* bp  = (const float*)d_in[6];
  const float* wu1 = (const float*)d_in[7];
  const float* bu1 = (const float*)d_in[8];
  const float* wu2 = (const float*)d_in[9];
  const float* bu2 = (const float*)d_in[10];

  int n_img = in_sizes[0] / 100;  // z is (N, 100, 1, 1)

  (void)hipFuncSetAttribute((const void*)nca_persist_kernel,
                            hipFuncAttributeMaxDynamicSharedMemorySize,
                            (int)SMEM_BYTES);

  nca_persist_kernel<<<n_img, 256, SMEM_BYTES, stream>>>(
      z, w1, b1, w2, b2, wp, bp, wu1, bu1, wu2, bu2, (float*)d_out);
}
